// MDLSTM_29171417874579
// MI455X (gfx1250) — matrix operand order repaired and checked by reading, not compiled, not run on hardware
//
#include <hip/hip_runtime.h>
#include <hip/hip_bf16.h>

// MD-LSTM (4-direction 2D LSTM) for MI455X / gfx1250.
//
// Design (see analysis): one fused wavefront kernel. Workgroup = (dir, row),
// 4*32 = 128 workgroups of 320 threads (10 waves). Per pixel each wave does
// 12x v_wmma_f32_16x16x32_bf16 (x@W + h_above@U0 + h_left@U1 for its 2
// N-tiles of the 320-wide gate vector), with ALL weight B-fragments resident
// in VGPRs for the entire kernel. Rows synchronize through agent-scope
// release/acquire progress flags (producer-consumer streaming wavefront).
// Scratch usage ~104 MB (xT bf16 + h ring bf16 + c ring f32 + flags).

#define BDIM 16
#define CDIM 64
#define HDIM 32
#define WDIM 128
#define ODIM 64
#define GDIM 320          // 5*O
#define NTHREADS 320      // 10 waves * 32

typedef __attribute__((ext_vector_type(16))) __bf16 v16bf;
typedef __attribute__((ext_vector_type(8)))  __bf16 v8bf;
typedef __attribute__((ext_vector_type(8)))  float  v8f;

__device__ __forceinline__ float fast_sigmoid(float x) {
  return 1.0f / (1.0f + __expf(-x));
}
__device__ __forceinline__ float fast_tanh(float x) {
  // 1 - 2/(e^{2x}+1): monotone, saturates correctly at +-1
  float e = __expf(2.0f * x);
  return 1.0f - 2.0f / (e + 1.0f);
}

// ---- WMMA fragment loaders (CDNA5 16-bit layouts, wave32) -----------------

// B fragment (32x16 K x N tile) from an f32 row-major [K][GDIM] weight matrix.
// Lane<16: N=n+lane,  K = kt*32 + {0..7, 16..23}; lane>=16: N=n+lane-16, K = kt*32 + {8..15, 24..31}.
__device__ __forceinline__ v16bf load_b_frag(const float* __restrict__ M,
                                             int kt, int nbase, int ln) {
  const int nn = nbase + (ln & 15);
  const int kb = kt * 32 + ((ln & 16) ? 8 : 0);
  v16bf r;
#pragma unroll
  for (int j = 0; j < 16; ++j)
    r[j] = (__bf16)M[(size_t)(kb + (j < 8 ? j : j + 8)) * GDIM + nn];
  return r;
}

// A fragment (16x32 M x K tile) from a bf16 row-major [16][64] block.
// Lane<16: M=lane, K = kt*32 + {0..7, 16..23}; lane>=16: M=lane-16, K = kt*32 + {8..15, 24..31}.
__device__ __forceinline__ v16bf load_a_frag(const __bf16* S, int kt, int ln) {
  const int row = ln & 15;
  const int kb  = kt * 32 + ((ln & 16) ? 8 : 0);
  const v8bf* p = (const v8bf*)(S + row * 64 + kb);
  v8bf lo = p[0];
  v8bf hi = p[2];   // +16 halves
  v16bf r;
#pragma unroll
  for (int j = 0; j < 8; ++j) { r[j] = lo[j]; r[8 + j] = hi[j]; }
  return r;
}

__device__ __forceinline__ v8f wmma_bf16(v16bf a, v16bf b, v8f c) {
  return __builtin_amdgcn_wmma_f32_16x16x32_bf16(false, a, false, b,
                                                 (short)0, c, false, false);
}

// ---- Kernels ---------------------------------------------------------------

__global__ void mdlstm_init(float* __restrict__ out, size_t n,
                            int* __restrict__ progress) {
  size_t i = (size_t)blockIdx.x * blockDim.x + threadIdx.x;
  if (i < n) out[i] = 0.0f;
  if (i < 4 * HDIM) progress[i] = 0;
}

// x (B,C,H,W) f32  ->  xT [(h*W+w)*16 + b][c] bf16 (contiguous 16x64 pixel tiles)
__global__ void mdlstm_xpose(const float* __restrict__ x,
                             __bf16* __restrict__ xT) {
  size_t i = (size_t)blockIdx.x * blockDim.x + threadIdx.x;
  const size_t N = (size_t)HDIM * WDIM * BDIM * CDIM;
  if (i >= N) return;
  int c = (int)(i & 63);
  int b = (int)((i >> 6) & 15);
  size_t hw = i >> 10;
  int w = (int)(hw & (WDIM - 1));
  int h = (int)(hw >> 7);
  float v = x[(((size_t)b * CDIM + c) * HDIM + h) * WDIM + w];
  xT[i] = (__bf16)v;
}

__global__ __launch_bounds__(NTHREADS, 1) void mdlstm_wavefront(
    const __bf16* __restrict__ xT,
    const float* __restrict__ ws, const float* __restrict__ u0s,
    const float* __restrict__ u1s, const float* __restrict__ bs,
    __bf16* __restrict__ hbuf, float* __restrict__ cbuf,
    int* __restrict__ progress,
    float* __restrict__ out_h, float* __restrict__ out_c) {
  const int tid = threadIdx.x;
  const int wv  = tid >> 5;
  const int ln  = tid & 31;
  const int blk = blockIdx.x;
  const int d   = blk >> 5;   // direction 0..3
  const int hh  = blk & 31;   // row in flipped frame

  const float* Wd  = ws  + (size_t)d * CDIM * GDIM;
  const float* U0d = u0s + (size_t)d * ODIM * GDIM;
  const float* U1d = u1s + (size_t)d * ODIM * GDIM;
  const float* Bd  = bs  + (size_t)d * GDIM;

  const int n0 = wv * 32;   // this wave owns gate columns [n0, n0+32)
  const int n1 = n0 + 16;

  // Weight B-fragments resident in registers for the whole kernel.
  v16bf bw[2][2], bu0[2][2], bu1[2][2];   // [k-tile][n-tile]
#pragma unroll
  for (int kt = 0; kt < 2; ++kt) {
    bw [kt][0] = load_b_frag(Wd,  kt, n0, ln);
    bw [kt][1] = load_b_frag(Wd,  kt, n1, ln);
    bu0[kt][0] = load_b_frag(U0d, kt, n0, ln);
    bu0[kt][1] = load_b_frag(U0d, kt, n1, ln);
    bu1[kt][0] = load_b_frag(U1d, kt, n0, ln);
    bu1[kt][1] = load_b_frag(U1d, kt, n1, ln);
  }
  const float bias0 = Bd[n0 + (ln & 15)];
  const float bias1 = Bd[n1 + (ln & 15)];

  __shared__ float  gateLDS[16 * GDIM];   // raw gate pre-activations (16 x 320)
  __shared__ __bf16 hLDS[16 * 64];        // h_left for next pixel
  __shared__ float  cLDS[16 * 64];        // c_left for next pixel

  const size_t rowpix   = ((size_t)d * HDIM + hh) * WDIM;              // this row's pixel base
  const size_t abovepix = (hh > 0) ? ((size_t)d * HDIM + hh - 1) * WDIM : 0;
  int* myflag = progress + d * HDIM + hh;
  int* upflag = (hh > 0) ? (progress + d * HDIM + hh - 1) : progress;

  for (int ww = 0; ww < WDIM; ++ww) {
    const int h = (d & 2) ? (HDIM - 1 - hh) : hh;   // global (un-flipped) coords
    const int w = (d & 1) ? (WDIM - 1 - ww) : ww;

    // Wait until the row above has produced pixel ww (streaming wavefront).
    if (hh > 0) {
      if (tid == 0) {
        while (__hip_atomic_load(upflag, __ATOMIC_ACQUIRE,
                                 __HIP_MEMORY_SCOPE_AGENT) <= ww) {
          __builtin_amdgcn_s_sleep(1);
        }
      }
      __syncthreads();
      __builtin_amdgcn_fence(__ATOMIC_ACQUIRE, "agent");
    }

    // --- GEMM phase: gates = bias + x@W (+ h_above@U0) (+ h_left@U1) ------
    const __bf16* xp = xT + ((size_t)h * WDIM + w) * (BDIM * CDIM);
    v16bf ax0 = load_a_frag(xp, 0, ln);
    v16bf ax1 = load_a_frag(xp, 1, ln);

    v8f acc0, acc1;
#pragma unroll
    for (int j = 0; j < 8; ++j) { acc0[j] = bias0; acc1[j] = bias1; }

    acc0 = wmma_bf16(ax0, bw[0][0], acc0);
    acc0 = wmma_bf16(ax1, bw[1][0], acc0);
    acc1 = wmma_bf16(ax0, bw[0][1], acc1);
    acc1 = wmma_bf16(ax1, bw[1][1], acc1);

    if (hh > 0) {   // uniform branch: EXEC stays all-ones at WMMA
      const __bf16* hap = hbuf + (abovepix + ww) * (BDIM * ODIM);
      v16bf ha0 = load_a_frag(hap, 0, ln);
      v16bf ha1 = load_a_frag(hap, 1, ln);
      acc0 = wmma_bf16(ha0, bu0[0][0], acc0);
      acc0 = wmma_bf16(ha1, bu0[1][0], acc0);
      acc1 = wmma_bf16(ha0, bu0[0][1], acc1);
      acc1 = wmma_bf16(ha1, bu0[1][1], acc1);
    }
    if (ww > 0) {   // uniform branch
      v16bf hl0 = load_a_frag(&hLDS[0], 0, ln);
      v16bf hl1 = load_a_frag(&hLDS[0], 1, ln);
      acc0 = wmma_bf16(hl0, bu1[0][0], acc0);
      acc0 = wmma_bf16(hl1, bu1[1][0], acc0);
      acc1 = wmma_bf16(hl0, bu1[0][1], acc1);
      acc1 = wmma_bf16(hl1, bu1[1][1], acc1);
    }

    // Spill raw gates to LDS (C/D layout: VGPR j -> M = j + 8*(lane>=16), N = lane&15).
    {
      const int r8  = (ln & 16) ? 8 : 0;
      const int nc0 = n0 + (ln & 15);
      const int nc1 = n1 + (ln & 15);
#pragma unroll
      for (int j = 0; j < 8; ++j) {
        gateLDS[(r8 + j) * GDIM + nc0] = acc0[j];
        gateLDS[(r8 + j) * GDIM + nc1] = acc1[j];
      }
    }
    __syncthreads();

    // --- Elementwise LSTM cell update (1024 (b,o) elements over 320 threads) ---
#pragma unroll
    for (int k = 0; k < 4; ++k) {
      int e = tid + k * NTHREADS;
      if (e < BDIM * ODIM) {
        int b = e >> 6, o = e & 63;
        const float* g = gateLDS + b * GDIM + o;
        float it = fast_sigmoid(g[0]);
        float ft = fast_sigmoid(g[64]);
        float gt = fast_tanh  (g[128]);
        float ot = fast_sigmoid(g[192]);
        float lt = fast_sigmoid(g[256]);
        float ca = (hh > 0) ? cbuf[(abovepix + ww) * (BDIM * ODIM) + e] : 0.0f;
        float cl = (ww > 0) ? cLDS[e] : 0.0f;
        float ct = ft * (lt * ca + (1.0f - lt) * cl) + it * gt;
        float ht = ot * fast_tanh(ct);
        cLDS[e] = ct;
        hLDS[e] = (__bf16)ht;
        hbuf[(rowpix + ww) * (BDIM * ODIM) + e] = (__bf16)ht;
        cbuf[(rowpix + ww) * (BDIM * ODIM) + e] = ct;
        size_t oi = (((size_t)b * ODIM + o) * HDIM + h) * WDIM + w;  // (B,O,H,W)
        atomicAdd(out_h + oi, 0.25f * ht);
        atomicAdd(out_c + oi, 0.25f * ct);
      }
    }
    __threadfence();      // make hbuf/cbuf stores agent-visible
    __syncthreads();
    if (tid == 0) {
      __hip_atomic_store(myflag, ww + 1, __ATOMIC_RELEASE,
                         __HIP_MEMORY_SCOPE_AGENT);
    }
  }
}

// ---- Host launch -----------------------------------------------------------

extern "C" void kernel_launch(void* const* d_in, const int* in_sizes, int n_in,
                              void* d_out, int out_size, void* d_ws, size_t ws_size,
                              hipStream_t stream) {
  (void)in_sizes; (void)n_in; (void)ws_size;
  const float* x   = (const float*)d_in[0];  // (16,64,32,128)
  const float* ws_ = (const float*)d_in[1];  // (4,64,320)
  const float* u0s = (const float*)d_in[2];  // (4,64,320)
  const float* u1s = (const float*)d_in[3];  // (4,64,320)
  const float* bs  = (const float*)d_in[4];  // (4,320)

  float* out_h = (float*)d_out;
  float* out_c = out_h + (size_t)BDIM * ODIM * HDIM * WDIM;

  // Scratch layout (total ~104 MB):
  char* wsb = (char*)d_ws;
  __bf16* xT   = (__bf16*)(wsb);                        //  8,388,608 B
  __bf16* hbuf = (__bf16*)(wsb + 8388608);              // 33,554,432 B
  float*  cbuf = (float*) (wsb + 8388608 + 33554432);   // 67,108,864 B
  int* progress = (int*)  (wsb + 8388608 + 33554432 + 67108864);  // 512 B

  const size_t outN = (size_t)2 * BDIM * ODIM * HDIM * WDIM;   // 8,388,608
  const size_t xtN  = (size_t)HDIM * WDIM * BDIM * CDIM;       // 4,194,304

  mdlstm_init<<<(unsigned)((outN + 255) / 256), 256, 0, stream>>>(
      (float*)d_out, outN, progress);
  mdlstm_xpose<<<(unsigned)((xtN + 255) / 256), 256, 0, stream>>>(x, xT);
  mdlstm_wavefront<<<4 * HDIM, NTHREADS, 0, stream>>>(
      xT, ws_, u0s, u1s, bs, hbuf, cbuf, progress, out_h, out_c);
  (void)out_size;
}
